// LegendreKANLayer_38405597561332
// MI455X (gfx1250) — compile-verified, matrix operand order checked by reading, not run on hardware
//
#include <hip/hip_runtime.h>
#include <hip/hip_bf16.h>
#include <math.h>

// ---------------------------------------------------------------------------
// LegendreKAN layer, fused for MI455X (gfx1250, wave32, WMMA bf16)
//   x      : (65536, 256) f32
//   c_basis: (64, 256, 8) f32    -> B matrix (K=2048, N=64)
//   bias   : (64,) f32
//   out    : (65536, 64) f32
// GEMM: M=65536, N=64, K=2048; A = Legendre basis computed in-register.
// B fragments are double-buffered in registers one K-chunk ahead so the
// WGP$/L2 load latency is hidden behind 4 WMMAs + the Legendre VALU chain.
// ---------------------------------------------------------------------------

typedef __attribute__((ext_vector_type(16))) __bf16 v16bf;
typedef __attribute__((ext_vector_type(2)))  __bf16 v2bf;
typedef __attribute__((ext_vector_type(8)))  float  v8f;

#define D_DEG     8
#define N_IN      256
#define N_OUT     64
#define K_TOTAL   (N_IN * D_DEG)      // 2048
#define KC_COUNT  (K_TOTAL / 32)      // 64 chunks of K=32
#define M_TILE    128
#define ROW_PITCH 258                 // floats; offsets {2i}∪{2i+1} mod 64 -> conflict-free
#define NTILES    (N_OUT / 16)        // 4

// P1..P8 via recurrence with reciprocal constants (no divides).
__device__ __forceinline__ void legendre8(float t, float* __restrict__ p) {
    const float ak[7] = {1.5f, 5.f/3.f, 1.75f, 1.8f, 11.f/6.f, 13.f/7.f, 1.875f};
    const float bk[7] = {0.5f, 2.f/3.f, 0.75f, 0.8f,  5.f/6.f,  6.f/7.f, 0.875f};
    float pm1 = 1.0f;   // P0
    float pc  = t;      // P1
    p[0] = pc;
#pragma unroll
    for (int k = 0; k < 7; ++k) {
        float pn = ak[k] * t * pc - bk[k] * pm1;
        p[k + 1] = pn;
        pm1 = pc;
        pc  = pn;
    }
}

// ---------------------------------------------------------------------------
// Repack c_basis (n_out, n_in, D) f32 -> bf16 B fragments in WMMA lane order.
// Layout: frag index j = ((kc*4 + t)*32 + lane)*8 + vgpr   (each entry = v2bf)
//   lane <  16 : N = t*16+lane,    K half-offset 0
//   lane >= 16 : N = t*16+lane-16, K half-offset 8
//   vgpr 0..3 -> K pairs {0,1},{2,3},{4,5},{6,7}; vgpr 4..7 -> +16
// ---------------------------------------------------------------------------
__global__ void pack_b_kernel(const float* __restrict__ cb, v2bf* __restrict__ bp) {
    int j = blockIdx.x * blockDim.x + threadIdx.x;
    if (j >= KC_COUNT * NTILES * 32 * 8) return;
    int vg   =  j        & 7;
    int lane = (j >> 3)  & 31;
    int t    = (j >> 8)  & 3;
    int kc   =  j >> 10;
    int n  = t * 16 + (lane & 15);
    int kb = kc * 32 + ((lane & 16) ? 8 : 0)
           + ((vg < 4) ? (2 * vg) : (16 + 2 * (vg - 4)));
    v2bf p;
    p.x = (__bf16)cb[n * K_TOTAL + kb];
    p.y = (__bf16)cb[n * K_TOTAL + kb + 1];
    bp[j] = p;
}

// ---------------------------------------------------------------------------
// Main fused kernel: 256 threads = 8 waves; each wave -> 16 rows x 64 cols.
// ---------------------------------------------------------------------------
__global__ __launch_bounds__(256, 2)
void kan_wmma_kernel(const float* __restrict__ x,
                     const v16bf* __restrict__ bp,
                     const float* __restrict__ bias,
                     float* __restrict__ out) {
    extern __shared__ float t_lds[];   // M_TILE * ROW_PITCH floats (tanh tile)

    const int tid       = threadIdx.x;
    const int wave      = tid >> 5;
    const int lane      = tid & 31;
    const int block_row = blockIdx.x * M_TILE;

    // Stage tanh(x) for the 128-row tile (coalesced, each element read once).
#pragma unroll 4
    for (int i = tid; i < M_TILE * N_IN; i += 256) {
        int r = i >> 8;              // / N_IN
        int c = i & (N_IN - 1);
        t_lds[r * ROW_PITCH + c] = tanhf(x[(block_row + r) * N_IN + c]);
    }
    __syncthreads();

    const int half16 = (lane & 16) ? 1 : 0;   // which K half this lane owns
    const int r16    = lane & 15;             // row within the 16x16 tile
    const float* trow = &t_lds[(wave * 16 + r16) * ROW_PITCH];

    v8f acc[NTILES] = {};

    // Prime the register double-buffer with chunk 0's B fragments.
    const v16bf* __restrict__ bbase = bp + lane;
    v16bf bcur[NTILES];
#pragma unroll
    for (int t = 0; t < NTILES; ++t) bcur[t] = bbase[t * 32];

#pragma unroll 2
    for (int kc = 0; kc < KC_COUNT; ++kc) {
        // --- A fragment: two features per lane, 8 Legendre degrees each. ---
        const int f = (kc << 2) + half16;
        float ta = trow[f];
        float tb = trow[f + 2];
        float pa[8], pb[8];
        legendre8(ta, pa);
        legendre8(tb, pb);

        v16bf a;
#pragma unroll
        for (int d = 0; d < 8; ++d) {
            a[d]     = (__bf16)pa[d];
            a[d + 8] = (__bf16)pb[d];
        }

        // --- Issue next chunk's B loads before consuming this chunk's. ---
        const int kn = (kc + 1) & (KC_COUNT - 1);           // wrap: branch-free
        const v16bf* __restrict__ bnextp = bbase + kn * (NTILES * 32);
        v16bf bnext[NTILES];
#pragma unroll
        for (int t = 0; t < NTILES; ++t) bnext[t] = bnextp[t * 32];

        // Warm WGP$ a few chunks ahead (global_prefetch_b8; no VGPR cost).
        __builtin_prefetch(bbase + ((kc + 4) & (KC_COUNT - 1)) * (NTILES * 32), 0, 3);

        // --- 4 WMMAs on the already-resident fragments. ---
#pragma unroll
        for (int t = 0; t < NTILES; ++t) {
            acc[t] = __builtin_amdgcn_wmma_f32_16x16x32_bf16(
                /*neg_a=*/false, a, /*neg_b=*/false, bcur[t],
                /*c_mod=*/(short)0, acc[t],
                /*reuse_a=*/false, /*reuse_b=*/false);
        }
#pragma unroll
        for (int t = 0; t < NTILES; ++t) bcur[t] = bnext[t];
    }

    // Store: C/D layout -> VGPR r: rows (r | r+8), N = lane&15 per half.
    const int col0    = lane & 15;
    const int rowbase = block_row + wave * 16 + (half16 ? 8 : 0);
#pragma unroll
    for (int t = 0; t < NTILES; ++t) {
        const int col = t * 16 + col0;
        const float bv = bias[col];
        v8f a = acc[t];
#pragma unroll
        for (int r = 0; r < 8; ++r) {
            out[(rowbase + r) * N_OUT + col] = a[r] + bv;
        }
    }
}

// ---------------------------------------------------------------------------
extern "C" void kernel_launch(void* const* d_in, const int* in_sizes, int n_in,
                              void* d_out, int out_size, void* d_ws, size_t ws_size,
                              hipStream_t stream) {
    const float* x    = (const float*)d_in[0];
    const float* cb   = (const float*)d_in[1];
    const float* bias = (const float*)d_in[2];
    float*       out  = (float*)d_out;

    const int batch = in_sizes[0] / N_IN;   // 65536

    // 1) Repack B into bf16 WMMA fragment order in workspace (256 KB).
    const int nfrag = KC_COUNT * NTILES * 32 * 8;
    pack_b_kernel<<<(nfrag + 255) / 256, 256, 0, stream>>>(cb, (v2bf*)d_ws);

    // 2) Fused basis + WMMA GEMM.
    const int    grid = batch / M_TILE;
    const size_t lds  = (size_t)M_TILE * ROW_PITCH * sizeof(float);
    kan_wmma_kernel<<<grid, 256, lds, stream>>>(x, (const v16bf*)d_ws, bias, out);
}
